// MP_DENSE_45440753992101
// MI455X (gfx1250) — compile-verified, matrix operand order checked by reading, not run on hardware
//
#include <hip/hip_runtime.h>
#include <hip/hip_bf16.h>

typedef __attribute__((ext_vector_type(16))) __bf16 v16bf;
typedef __attribute__((ext_vector_type(8)))  __bf16 v8bf;
typedef __attribute__((ext_vector_type(4)))  __bf16 v4bf;
typedef __attribute__((ext_vector_type(8)))  float  v8f;

#define HSIZE 128

__device__ __forceinline__ void lds_fence() {
  asm volatile("s_wait_dscnt 0x0" ::: "memory");
}

// Stage 16 rows x 128 f32 (row-major) -> bf16 LDS tile [16][128], coalesced.
__device__ __forceinline__ void stage_tile(const float* __restrict__ src,
                                           __bf16* dst, int lane) {
#pragma unroll
  for (int i = 0; i < 16; ++i) {
    int linear = i * 32 + lane;          // float4 index within 16x128 tile
    int row = linear >> 5;
    int c4  = linear & 31;
    float4 v = *(const float4*)(src + row * HSIZE + c4 * 4);
    v4bf o = {(__bf16)v.x, (__bf16)v.y, (__bf16)v.z, (__bf16)v.w};
    *(v4bf*)(dst + row * HSIZE + c4 * 4) = o;      // ds_store_b64
  }
}

// A fragment (16x32 bf16) for k-tile kt from LDS tile [16][128].
// ISA layout: lane m (0-15): elems 0-7 = K 0..7, elems 8-15 = K 16..23;
//             lane m+16:     elems 0-7 = K 8..15, elems 8-15 = K 24..31.
__device__ __forceinline__ v16bf load_afrag(const __bf16* tile, int lane, int kt) {
  int m = lane & 15, half = lane >> 4;
  const __bf16* p = tile + m * HSIZE + kt * 32 + half * 8;
  v8bf lo = *(const v8bf*)p;              // ds_load_b128
  v8bf hi = *(const v8bf*)(p + 16);       // ds_load_b128
  return __builtin_shufflevector(lo, hi, 0, 1, 2, 3, 4, 5, 6, 7,
                                 8, 9, 10, 11, 12, 13, 14, 15);
}

// One GEMM phase: D[16x128] = act( A[16x(KT*32)] @ W + bias ).
// W fragments pre-shuffled in global memory: [nt][kt][lane][16] bf16.
// Accumulators start at 0 so the first v_wmma can take SRC2 = inline 0;
// bias is fused into the epilogue.
// C-layout writeback: VGPR i -> row (lane<16 ? i : 8+i), col = nt*16 + (lane&15).
template <int KT, bool RELU, bool TO_LDS>
__device__ __forceinline__ void gemm_phase(const v16bf* A0, const v16bf* A1,
                                           const __bf16* __restrict__ wf,
                                           const float* __restrict__ bias,
                                           __bf16* dstLds,
                                           float* __restrict__ dstGlb, int lane) {
  int n0 = lane & 15, mhi = (lane >> 4) * 8;
  for (int nt = 0; nt < 8; ++nt) {
    float bv = bias[nt * 16 + n0];
    v8f acc = {};                          // inline-0 C for first WMMA
#pragma unroll
    for (int kt = 0; kt < KT; ++kt) {
      v16bf b = *(const v16bf*)(wf + (((nt * KT + kt) * 32 + lane) << 4));
      v16bf a = (kt < 4) ? A0[kt] : A1[kt - 4];
      acc = __builtin_amdgcn_wmma_f32_16x16x32_bf16(false, a, false, b,
                                                    (short)0, acc, false, false);
    }
#pragma unroll
    for (int i = 0; i < 8; ++i) {
      float v = acc[i] + bv;
      if (RELU) v = v > 0.f ? v : 0.f;
      int m = mhi + i;
      if (TO_LDS) dstLds[m * HSIZE + nt * 16 + n0] = (__bf16)v;
      else        dstGlb[(long long)m * HSIZE + nt * 16 + n0] = v;
    }
  }
}

// Pre-shuffle all four weight matrices (f32 row-major KxN) into bf16
// WMMA B-fragment order in d_ws: [nt][kt][lane][16].
__global__ __launch_bounds__(256) void prep_weights_kernel(
    const float* __restrict__ Wm1, const float* __restrict__ Wm2,
    const float* __restrict__ Wu1, const float* __restrict__ Wu2,
    __bf16* __restrict__ wf) {
  int tid = blockIdx.x * 256 + threadIdx.x;
  if (tid >= 98304) return;                 // (64+32+64+32)*512 elems
  const float* W;
  int KT, local;
  if (tid < 32768)      { W = Wm1; KT = 8; local = tid; }
  else if (tid < 49152) { W = Wm2; KT = 4; local = tid - 32768; }
  else if (tid < 81920) { W = Wu1; KT = 8; local = tid - 49152; }
  else                  { W = Wu2; KT = 4; local = tid - 81920; }
  int e    = local & 15;
  int lane = (local >> 4) & 31;
  int rest = local >> 9;
  int kt   = rest % KT;
  int nt   = rest / KT;
  int n    = nt * 16 + (lane & 15);
  int half = lane >> 4;
  int k    = kt * 32 + half * 8 + (e < 8 ? e : 16 + (e - 8));
  wf[tid] = (__bf16)W[k * HSIZE + n];
}

__global__ __launch_bounds__(128) void mp_dense_kernel(
    const float* __restrict__ hself, const float* __restrict__ hother,
    const __bf16* __restrict__ wf,
    const float* __restrict__ bm1, const float* __restrict__ bm2,
    const float* __restrict__ bu1, const float* __restrict__ bu2,
    float* __restrict__ out) {
  __shared__ __align__(16) __bf16 sbuf[4][2][16 * HSIZE];   // 32 KB
  int lane = threadIdx.x & 31;
  int wave = threadIdx.x >> 5;
  long long r0 = (long long)blockIdx.x * 64 + wave * 16;

  const __bf16* wm1f = wf;
  const __bf16* wm2f = wm1f + 8 * 8 * 32 * 16;
  const __bf16* wu1f = wm2f + 8 * 4 * 32 * 16;
  const __bf16* wu2f = wu1f + 8 * 8 * 32 * 16;

  __bf16* b0 = sbuf[wave][0];
  __bf16* b1 = sbuf[wave][1];

  const float* xs = hself + r0 * HSIZE;
  const float* xo = hother + r0 * HSIZE;
  __builtin_prefetch(xo, 0, 0);            // global_prefetch_b8

  v16bf Axs[4], Atmp[4];

  // Stage hself rows, capture its A fragments (kept live all kernel).
  stage_tile(xs, b0, lane);
  lds_fence();
#pragma unroll
  for (int kt = 0; kt < 4; ++kt) Axs[kt] = load_afrag(b0, lane, kt);
  lds_fence();
  // Stage hother rows into same buffer, capture fragments.
  stage_tile(xo, b0, lane);
  lds_fence();
#pragma unroll
  for (int kt = 0; kt < 4; ++kt) Atmp[kt] = load_afrag(b0, lane, kt);
  lds_fence();

  // phase 1: h1 = relu([xs,xo] @ Wm1 + bm1) -> b1
  gemm_phase<8, true, true>(Axs, Atmp, wm1f, bm1, b1, nullptr, lane);
  lds_fence();
#pragma unroll
  for (int kt = 0; kt < 4; ++kt) Atmp[kt] = load_afrag(b1, lane, kt);
  lds_fence();

  // phase 2: msg = h1 @ Wm2 + bm2 -> b0
  gemm_phase<4, false, true>(Atmp, Atmp, wm2f, bm2, b0, nullptr, lane);
  lds_fence();
#pragma unroll
  for (int kt = 0; kt < 4; ++kt) Atmp[kt] = load_afrag(b0, lane, kt);
  lds_fence();

  // phase 3: h2 = relu([xs,msg] @ Wu1 + bu1) -> b1
  gemm_phase<8, true, true>(Axs, Atmp, wu1f, bu1, b1, nullptr, lane);
  lds_fence();
#pragma unroll
  for (int kt = 0; kt < 4; ++kt) Atmp[kt] = load_afrag(b1, lane, kt);
  lds_fence();

  // phase 4: out = h2 @ Wu2 + bu2 -> global (f32)
  gemm_phase<4, false, false>(Atmp, Atmp, wu2f, bu2, nullptr,
                              out + r0 * HSIZE, lane);
}

extern "C" void kernel_launch(void* const* d_in, const int* in_sizes, int n_in,
                              void* d_out, int out_size, void* d_ws, size_t ws_size,
                              hipStream_t stream) {
  const float* hself  = (const float*)d_in[0];
  const float* hother = (const float*)d_in[1];
  const float* Wm1 = (const float*)d_in[2];
  const float* bm1 = (const float*)d_in[3];
  const float* Wm2 = (const float*)d_in[4];
  const float* bm2 = (const float*)d_in[5];
  const float* Wu1 = (const float*)d_in[6];
  const float* bu1 = (const float*)d_in[7];
  const float* Wu2 = (const float*)d_in[8];
  const float* bu2 = (const float*)d_in[9];
  float* out = (float*)d_out;
  __bf16* wf = (__bf16*)d_ws;               // needs 192 KB scratch

  int E = in_sizes[0] / HSIZE;              // 640000 rows

  prep_weights_kernel<<<(98304 + 255) / 256, 256, 0, stream>>>(Wm1, Wm2, Wu1,
                                                               Wu2, wf);
  mp_dense_kernel<<<E / 64, 128, 0, stream>>>(hself, hother, wf, bm1, bm2, bu1,
                                              bu2, out);
}
